// DGM_d_48550310314077
// MI455X (gfx1250) — compile-verified, matrix operand order checked by reading, not run on hardware
//
#include <hip/hip_runtime.h>
#include <hip/hip_bf16.h>
#include <math.h>

// ---------------------------------------------------------------------------
// kNN graph (DGM euclidean) for MI455X / gfx1250.
// Gram matrix via split-bf16 WMMA (v_wmma_f32_16x16x32_bf16), B tiles staged
// through LDS with GLOBAL_LOAD_ASYNC_TO_LDS_B128 (double-buffered, ASYNCcnt),
// fused per-row top-10 selection, then edge/logprob emission.
// ---------------------------------------------------------------------------

#define NPTS   8192
#define DDIM   256
#define KNN    10
#define CSPLIT 8                      // column slices (grid.y) for parallelism
#define CCOLS  (NPTS / CSPLIT)        // 1024 columns per slice
#define CTILES (CCOLS / 16)           // 64 column tiles per slice
#define BPAD   264                    // 256 + 8 bf16 pad: conflict-free LDS rows
#define FINF   3.402823e38f

typedef __attribute__((ext_vector_type(8)))  __bf16 v8bf;
typedef __attribute__((ext_vector_type(16))) __bf16 v16bf;
typedef __attribute__((ext_vector_type(8)))  float  v8f;

__device__ __forceinline__ v8f wmma_bf16(v16bf a, v16bf b, v8f c) {
  // 8 args: (neg_a, A, neg_b, B, c_mod, C, reuse_a, reuse_b)
  return __builtin_amdgcn_wmma_f32_16x16x32_bf16(false, a, false, b,
                                                 (short)0, c, false, false);
}

// Async 16-byte Global -> LDS copy (per-lane addresses, tracked by ASYNCcnt).
__device__ __forceinline__ void async_b128(unsigned int ldsoff, const void* g) {
  asm volatile("global_load_async_to_lds_b128 %0, %1, off"
               :: "v"(ldsoff), "v"((unsigned long long)(uintptr_t)g)
               : "memory");
}
__device__ __forceinline__ void wait_async0() {
  asm volatile("s_wait_asynccnt 0" ::: "memory");
}
__device__ __forceinline__ void wait_ds0() {
  asm volatile("s_wait_dscnt 0" ::: "memory");
}

// Fragment layout for 16-bit A (16x32) / B (32x16) on gfx1250 (wave32):
// lane L -> row (L&15); K elements are two contiguous 8-element chunks at
// K = c0..c0+7 and K = c0+16..c0+23 with c0 = 8*(L>>4).  Both A (rows of X)
// and B (X^T columns == rows of X for a Gram matrix) reduce to "lane reads
// two 16-byte chunks of one row", so one loader serves global and LDS.
__device__ __forceinline__ v16bf load_frag(const __bf16* p) {
  v8bf a = *(const v8bf*)p;
  v8bf b = *(const v8bf*)(p + 16);
  v16bf r;
#pragma unroll
  for (int i = 0; i < 8; ++i) { r[i] = a[i]; r[i + 8] = b[i]; }
  return r;
}

// Branch-free sorted insert into ascending 10-list held in registers.
__device__ __forceinline__ void insert10(float v, int j,
                                         float (&bv)[KNN], int (&bi)[KNN]) {
  if (v < bv[KNN - 1]) {
#pragma unroll
    for (int t = KNN - 1; t >= 1; --t) {
      bool shift = v < bv[t - 1];
      bool place = (v < bv[t]) && !shift;
      float nv = shift ? bv[t - 1] : (place ? v : bv[t]);
      int   ni = shift ? bi[t - 1] : (place ? j : bi[t]);
      bv[t] = nv; bi[t] = ni;
    }
    if (v < bv[0]) { bv[0] = v; bi[0] = j; }
  }
}

// ---------------------------------------------------------------------------
// Kernel 1: copy x to output, split f32 -> bf16 hi/lo, row squared norms.
// ---------------------------------------------------------------------------
__global__ __launch_bounds__(256)
void prep_kernel(const float* __restrict__ x, float* __restrict__ xout,
                 __bf16* __restrict__ xhi, __bf16* __restrict__ xlo,
                 float* __restrict__ sq) {
  int row = blockIdx.x;
  int t   = threadIdx.x;
  float v = x[row * DDIM + t];
  xout[row * DDIM + t] = v;
  __bf16 h = (__bf16)v;
  xhi[row * DDIM + t] = h;
  xlo[row * DDIM + t] = (__bf16)(v - (float)h);

  __shared__ float red[256];
  red[t] = v * v;
  __syncthreads();
#pragma unroll
  for (int s = 128; s > 0; s >>= 1) {
    if (t < s) red[t] += red[t + s];
    __syncthreads();
  }
  if (t == 0) sq[row] = red[0];
}

// ---------------------------------------------------------------------------
// Kernel 2: Gram tiles via split-bf16 WMMA + fused per-row top-10.
// Block = 8 waves; each wave owns one 16-row tile (A hi/lo preloaded in
// VGPRs, full K=256).  The block cooperatively async-stages each 16-row B
// tile (hi+lo, 16KB) into double-buffered LDS, so each tile is fetched from
// L2 once per block instead of once per wave (8x traffic cut).
// Selection metric: sq[j] - 2*gram  (row term sq[i] is constant per row).
// ---------------------------------------------------------------------------
__global__ __launch_bounds__(256)
void gram_topk_kernel(const __bf16* __restrict__ xhi,
                      const __bf16* __restrict__ xlo,
                      const float* __restrict__ sq,
                      float* __restrict__ slice_v,  // [CSPLIT][NPTS][KNN]
                      int*   __restrict__ slice_i) {
  __shared__ __align__(16) __bf16 bHi[2][16][BPAD];
  __shared__ __align__(16) __bf16 bLo[2][16][BPAD];
  __shared__ float shTile[8][16][17];   // padded: kills bank conflicts
  __shared__ float shV[8][16][KNN];
  __shared__ int   shI[8][16][KNN];

  const int tid  = threadIdx.x;
  const int w    = tid >> 5;
  const int lane = tid & 31;
  const int m0   = (blockIdx.x * 8 + w) * 16;     // row-tile base
  const int slice  = blockIdx.y;
  const int jbegin = slice * CCOLS;

  const int lrow = lane & 15;          // fragment row within tile
  const int c0   = (lane >> 4) * 8;    // K-chunk base for this half-wave

  // Preload A fragments: 8 K-chunks x (hi, lo).
  v16bf ahi[8], alo[8];
  {
    const int arow = m0 + lrow;
#pragma unroll
    for (int kc = 0; kc < 8; ++kc) {
      const int base = arow * DDIM + kc * 32 + c0;
      ahi[kc] = load_frag(xhi + base);
      alo[kc] = load_frag(xlo + base);
    }
  }

  // Cooperative async stage of B tile jt into buffer buf.
  // 16 rows x 512B = 32 16B-chunks/row = 512 chunks; 2 chunks/thread/matrix.
  auto stage = [&](int buf, int jt) {
    const int j0 = jbegin + jt * 16;
#pragma unroll
    for (int q = 0; q < 2; ++q) {
      const int c   = 2 * tid + q;     // 0..511
      const int row = c >> 5;
      const int ch  = c & 31;          // 16B chunk within row
      const size_t goff = (size_t)(j0 + row) * DDIM + ch * 8;
      async_b128((unsigned int)(uintptr_t)&bHi[buf][row][ch * 8], xhi + goff);
      async_b128((unsigned int)(uintptr_t)&bLo[buf][row][ch * 8], xlo + goff);
    }
  };

  float bv[KNN]; int bi[KNN];
#pragma unroll
  for (int t = 0; t < KNN; ++t) { bv[t] = FINF; bi[t] = 0; }

  stage(0, 0);
  wait_async0();
  __syncthreads();

  for (int jt = 0; jt < CTILES; ++jt) {
    const int buf = jt & 1;
    if (jt + 1 < CTILES) stage(buf ^ 1, jt + 1);  // overlap next tile DMA

    const int j0 = jbegin + jt * 16;
    v8f acc = {};
#pragma unroll
    for (int kc = 0; kc < 8; ++kc) {
      v16bf bhi = load_frag(&bHi[buf][lrow][kc * 32 + c0]);
      v16bf blo = load_frag(&bLo[buf][lrow][kc * 32 + c0]);
      // split-bf16 fp32 emulation: hi*hi + hi*lo + lo*hi
      acc = wmma_bf16(ahi[kc], bhi, acc);
      acc = wmma_bf16(ahi[kc], blo, acc);
      acc = wmma_bf16(alo[kc], bhi, acc);
    }

    // C/D layout: lane holds column n=(L&15), rows r+8*(L>>4) in acc[r].
    const float sqj = sq[j0 + lrow];
#pragma unroll
    for (int r = 0; r < 8; ++r) {
      const int m = r + 8 * (lane >> 4);
      shTile[w][m][lrow] = sqj - 2.0f * acc[r];
    }
    wait_ds0();  // same-wave LDS ordering before cross-lane reads

    // Transposed read: lane owns row (L&15), columns c0..c0+7 of this tile.
#pragma unroll
    for (int c = 0; c < 8; ++c) {
      const int col = c0 + c;
      insert10(shTile[w][lrow][col], j0 + col, bv, bi);
    }

    wait_async0();     // next buffer fully landed in LDS
    __syncthreads();   // all waves done with current buffer + see new one
  }

  // Merge the two half-wave partial lists per row (cols 0..7 vs 8..15).
  if (lane >= 16) {
#pragma unroll
    for (int t = 0; t < KNN; ++t) { shV[w][lrow][t] = bv[t]; shI[w][lrow][t] = bi[t]; }
  }
  wait_ds0();
  if (lane < 16) {
#pragma unroll
    for (int t = 0; t < KNN; ++t) insert10(shV[w][lrow][t], shI[w][lrow][t], bv, bi);
    const int row = m0 + lrow;
    float* sv = slice_v + ((size_t)slice * NPTS + row) * KNN;
    int*   si = slice_i + ((size_t)slice * NPTS + row) * KNN;
#pragma unroll
    for (int t = 0; t < KNN; ++t) { sv[t] = bv[t]; si[t] = bi[t]; }
  }
}

// ---------------------------------------------------------------------------
// Kernel 3: merge the 8 per-slice top-10 lists into the final top-10.
// ---------------------------------------------------------------------------
__global__ __launch_bounds__(256)
void merge_kernel(const float* __restrict__ slice_v,
                  const int*   __restrict__ slice_i,
                  int* __restrict__ topi) {
  int row = blockIdx.x * blockDim.x + threadIdx.x;
  if (row >= NPTS) return;
  float bv[KNN]; int bi[KNN];
#pragma unroll
  for (int t = 0; t < KNN; ++t) { bv[t] = FINF; bi[t] = 0; }
  for (int s = 0; s < CSPLIT; ++s) {
    const float* sv = slice_v + ((size_t)s * NPTS + row) * KNN;
    const int*   si = slice_i + ((size_t)s * NPTS + row) * KNN;
#pragma unroll
    for (int t = 0; t < KNN; ++t) insert10(sv[t], si[t], bv, bi);
  }
#pragma unroll
  for (int t = 0; t < KNN; ++t) topi[row * KNN + t] = bi[t];
}

// ---------------------------------------------------------------------------
// Kernel 4: edges + logprobs.  One wave per flat edge p in [0, N*K).
// Reference flat-memory layout: edge pair p -> (topi[p%N][p/N], p/K), and
// logprobs[p] = -scale * || x[topi[p%N][p/N]] - x[p/K] ||^2  (exact f32).
// ---------------------------------------------------------------------------
__global__ __launch_bounds__(256)
void finalize_kernel(const float* __restrict__ x,
                     const int* __restrict__ topi,
                     const float* __restrict__ temperature,
                     float* __restrict__ out_edges,
                     float* __restrict__ out_lp) {
  const int w    = threadIdx.x >> 5;
  const int lane = threadIdx.x & 31;
  const int p    = blockIdx.x * 8 + w;
  if (p >= NPTS * KNN) return;

  const int nb   = topi[(p % NPTS) * KNN + (p / NPTS)];
  const int self = p / KNN;

  const float* xa = x + (size_t)nb * DDIM;
  const float* xb = x + (size_t)self * DDIM;
  float s = 0.0f;
#pragma unroll
  for (int i = 0; i < DDIM / 32; ++i) {
    const float d = xa[lane + i * 32] - xb[lane + i * 32];
    s += d * d;
  }
#pragma unroll
  for (int off = 16; off >= 1; off >>= 1) s += __shfl_xor(s, off, 32);

  if (lane == 0) {
    float t = temperature[0];
    t = fminf(fmaxf(t, -5.0f), 5.0f);
    const float scale = expf(t);
    out_lp[p] = -s * scale;
    out_edges[2 * p + 0] = (float)nb;
    out_edges[2 * p + 1] = (float)self;
  }
}

// ---------------------------------------------------------------------------
extern "C" void kernel_launch(void* const* d_in, const int* in_sizes, int n_in,
                              void* d_out, int out_size, void* d_ws, size_t ws_size,
                              hipStream_t stream) {
  (void)in_sizes; (void)n_in; (void)out_size; (void)ws_size;
  const float* x    = (const float*)d_in[0];
  const float* temp = (const float*)d_in[1];

  // Workspace carve-up (bytes), all 16B-aligned:
  char* ws = (char*)d_ws;
  __bf16* xhi = (__bf16*)(ws);                                    // 4 MB
  __bf16* xlo = (__bf16*)(ws + (size_t)NPTS * DDIM * 2);          // 4 MB
  float*  sq  = (float*)(ws + (size_t)NPTS * DDIM * 4);           // 32 KB
  float*  sv  = (float*)(ws + (size_t)NPTS * DDIM * 4 + NPTS * 4);
  int*    si  = (int*)((char*)sv + (size_t)CSPLIT * NPTS * KNN * 4);
  int*    ti  = (int*)((char*)si + (size_t)CSPLIT * NPTS * KNN * 4);

  float* out    = (float*)d_out;
  float* out_x  = out;                               // N*D
  float* out_e  = out + (size_t)NPTS * DDIM;         // 2*N*K
  float* out_lp = out_e + (size_t)2 * NPTS * KNN;    // N*K

  prep_kernel<<<NPTS, 256, 0, stream>>>(x, out_x, xhi, xlo, sq);

  dim3 ggrid(NPTS / 128, CSPLIT);                    // 8 row-tiles/block
  gram_topk_kernel<<<ggrid, 256, 0, stream>>>(xhi, xlo, sq, sv, si);

  merge_kernel<<<(NPTS + 255) / 256, 256, 0, stream>>>(sv, si, ti);

  finalize_kernel<<<(NPTS * KNN + 7) / 8, 256, 0, stream>>>(x, ti, temp,
                                                            out_e, out_lp);
}
